// KhopfieldAttention_18717467476127
// MI455X (gfx1250) — matrix operand order repaired and checked by reading, not compiled, not run on hardware
//
#include <hip/hip_runtime.h>
#include <hip/hip_bf16.h>

// ---------------------------------------------------------------------------
// KhopfieldAttention for MI455X (gfx1250, wave32, WMMA)
//   B=4, SEQ=1024, DIM=512, HEADS=8, DIM_HEAD=64, N_ITER=200
// Pipeline:
//   k_ln   : layernorm -> xn (bf16)
//   k_qkv  : xn @ w_qkv via v_wmma_f32_16x16x32_bf16; each wave produces one
//            q/k/v 16x16 tile triple (wave-uniform store routing, A reused x3)
//   k_dots : q @ k^T via WMMA, 16x64 tile per wave (A reused x4)
//   k_lml  : per-row 8-way concurrent bisection (native exp2/rcp) + fused attn@v
//   k_out  : headout @ w_out + b_out via WMMA, 16x64 tile per wave
// ---------------------------------------------------------------------------

typedef __attribute__((ext_vector_type(16))) __bf16 v16bf;
typedef __attribute__((ext_vector_type(8)))  __bf16 v8bf;
typedef __attribute__((ext_vector_type(8)))  float  v8f;

#define SEQ_ 1024
#define DIM_ 512
#define DH_  64
#define NH_  8

__device__ inline v8f wmma_bf16(v16bf a, v16bf b, v8f c) {
  // D = A(16x32) * B(32x16) + C(16x16), f32 accumulate
  return __builtin_amdgcn_wmma_f32_16x16x32_bf16(
      /*neg_a=*/false, a, /*neg_b=*/false, b,
      /*c_mod=*/(short)0, c, /*reuse_a=*/false, /*reuse_b=*/false);
}

// A fragment: 16x32 bf16, row-major source with leading dim `ld`.
// Lane l (0..15): M=l, K in {kb..kb+15}; lanes 16..31: kb=16.
__device__ inline v16bf load_a_bf(const __bf16* A, int ld, int r0, int k0, int lane) {
  const int m  = lane & 15;
  const int kb = (lane >> 4) * 16;
  const __bf16* base = A + (size_t)(r0 + m) * ld + k0 + kb;
  v8bf lo = *reinterpret_cast<const v8bf*>(base);       // K = kb .. kb+7
  v8bf hi = *reinterpret_cast<const v8bf*>(base + 8);   // K = kb+8 .. kb+15
  v16bf a;
#pragma unroll
  for (int i = 0; i < 8; ++i) { a[i] = lo[i]; a[8 + i] = hi[i]; }
  return a;
}

// B fragment: 32x16 from a row-major f32 [K x N] matrix (on-the-fly bf16 cvt).
// Lane l: N=l&15, K = (l>>4)*16 + i.
__device__ inline v16bf load_b_f32(const float* B, int ld, int k0, int n0, int lane) {
  const int n  = lane & 15;
  const int kb = (lane >> 4) * 16;
  v16bf b;
#pragma unroll
  for (int i = 0; i < 16; ++i)
    b[i] = (__bf16)B[(size_t)(k0 + kb + i) * ld + n0 + n];
  return b;
}

// B fragment for q@k^T: B(kk,n) = Kmat[n][kk] -> contiguous 16 bf16 per lane.
__device__ inline v16bf load_b_kt(const __bf16* K, int n0, int k0, int lane) {
  const int n  = lane & 15;
  const int kb = (lane >> 4) * 16;
  return *reinterpret_cast<const v16bf*>(K + (size_t)(n0 + n) * DH_ + k0 + kb);
}

// ------------------------------- layernorm ---------------------------------
__global__ __launch_bounds__(256) void k_ln(const float* __restrict__ x,
                                            const float* __restrict__ g,
                                            const float* __restrict__ bta,
                                            __bf16* __restrict__ xn) {
  const int row = blockIdx.x;           // 0..4095
  const int tid = threadIdx.x;
  __shared__ float rs[256], rs2[256];
  const float v0 = x[(size_t)row * DIM_ + tid];
  const float v1 = x[(size_t)row * DIM_ + 256 + tid];
  rs[tid]  = v0 + v1;
  rs2[tid] = v0 * v0 + v1 * v1;
  for (int s = 128; s; s >>= 1) {
    __syncthreads();
    if (tid < s) { rs[tid] += rs[tid + s]; rs2[tid] += rs2[tid + s]; }
  }
  __syncthreads();
  const float mu  = rs[0] * (1.0f / DIM_);
  const float var = rs2[0] * (1.0f / DIM_) - mu * mu;
  const float r   = rsqrtf(var + 1e-5f);
  xn[(size_t)row * DIM_ + tid]       = (__bf16)((v0 - mu) * r * g[tid]       + bta[tid]);
  xn[(size_t)row * DIM_ + 256 + tid] = (__bf16)((v1 - mu) * r * g[256 + tid] + bta[256 + tid]);
}

// ----------------------------- QKV projection ------------------------------
// [4096,512] @ [512,192]. Each wave: one m-tile x one n-offset, producing the
// q, k and v 16x16 tiles at columns (tn*16, 64+tn*16, 128+tn*16).
// A fragment loaded once per k-step and fed to 3 WMMAs; stores are uniform.
__global__ __launch_bounds__(256) void k_qkv(const __bf16* __restrict__ xn,
                                             const float*  __restrict__ wqkv,
                                             __bf16* __restrict__ qo,
                                             __bf16* __restrict__ ko,
                                             float*  __restrict__ vo) {
  const int wid  = threadIdx.x >> 5, lane = threadIdx.x & 31;
  const int wg   = blockIdx.x * 8 + wid;        // < 1024
  const int tm   = wg >> 2;                     // 0..255
  const int tn   = wg & 3;                      // 0..3
  const int r0   = tm * 16;
  v8f aq = {}, ak = {}, av = {};
  for (int k0 = 0; k0 < DIM_; k0 += 32) {
    v16bf a = load_a_bf(xn, DIM_, r0, k0, lane);
    aq = wmma_bf16(a, load_b_f32(wqkv, 3 * DH_, k0, tn * 16,            lane), aq);
    ak = wmma_bf16(a, load_b_f32(wqkv, 3 * DH_, k0, DH_ + tn * 16,      lane), ak);
    av = wmma_bf16(a, load_b_f32(wqkv, 3 * DH_, k0, 2 * DH_ + tn * 16,  lane), av);
  }
  const int n = lane & 15, mb = (lane >> 4) * 8;
#pragma unroll
  for (int r = 0; r < 8; ++r) {
    const size_t idx = (size_t)(r0 + mb + r) * DH_ + tn * 16 + n;
    qo[idx] = (__bf16)(aq[r] * 0.125f);         // fold DIM_HEAD^-0.5
    ko[idx] = (__bf16)ak[r];
    vo[idx] = av[r];
  }
}

// ------------------------------- dots = q k^T ------------------------------
// 16x64 tile per wave: 4 accumulators sharing one A fragment per k-step.
__global__ __launch_bounds__(256) void k_dots(const __bf16* __restrict__ q,
                                              const __bf16* __restrict__ k,
                                              float* __restrict__ dots) {
  const int wid  = threadIdx.x >> 5, lane = threadIdx.x & 31;
  const int wg   = blockIdx.x * 8 + wid;        // < 4096
  const int b    = wg >> 10;
  const int rem  = wg & 1023;
  const int r0   = b * SEQ_ + (rem >> 4) * 16;  // global q row
  const int c0   = (rem & 15) * 64;             // key col in batch
  const __bf16* kbase = k + (size_t)b * SEQ_ * DH_;
  v8f acc[4] = {};
#pragma unroll
  for (int k0 = 0; k0 < DH_; k0 += 32) {
    v16bf a = load_a_bf(q, DH_, r0, k0, lane);
#pragma unroll
    for (int t = 0; t < 4; ++t)
      acc[t] = wmma_bf16(a, load_b_kt(kbase, c0 + t * 16, k0, lane), acc[t]);
  }
  const int n = lane & 15, mb = (lane >> 4) * 8;
#pragma unroll
  for (int t = 0; t < 4; ++t)
#pragma unroll
    for (int r = 0; r < 8; ++r)
      dots[(size_t)(r0 + mb + r) * SEQ_ + c0 + t * 16 + n] = acc[t][r];
}

// ------------------- LML bisection (x8 concurrent) + attn@v ----------------
// One block per query row; wave w solves N = w+1. 200 iters; inner loop is
// add -> v_exp_f32 -> add -> v_rcp_f32 -> add (xs[t] = -x*log2e hoisted).
__global__ __launch_bounds__(256) void k_lml(const float* __restrict__ dots,
                                             const float* __restrict__ v,
                                             __bf16* __restrict__ outh) {
  const int row  = blockIdx.x;                  // b*1024 + i
  const int b    = row >> 10;
  const int tid  = threadIdx.x;
  const int wid  = tid >> 5, lane = tid & 31;

  __shared__ float x_lds[SEQ_];                 //  4 KB
  __shared__ float p_lds[NH_][SEQ_];            // 32 KB
  __shared__ float v_lds[64 * DH_];             // 16 KB
  __shared__ float red[256], red2[256];         //  2 KB

  // Load row, row-wide max/min for bisection bracket.
  float mx = -3.4e38f, mn = 3.4e38f;
#pragma unroll
  for (int q = 0; q < 4; ++q) {
    const float d = dots[(size_t)row * SEQ_ + tid + q * 256];
    x_lds[tid + q * 256] = d;
    mx = fmaxf(mx, d); mn = fminf(mn, d);
  }
  red[tid] = mx; red2[tid] = mn;
  for (int s = 128; s; s >>= 1) {
    __syncthreads();
    if (tid < s) { red[tid] = fmaxf(red[tid], red[tid + s]);
                   red2[tid] = fminf(red2[tid], red2[tid + s]); }
  }
  __syncthreads();
  float lo = -red[0] - 20.0f;
  float hi = -red2[0] + 20.0f;

  const float L2E = 1.44269504088896340736f;
  float xs[32];                                 // pre-negated, pre-scaled
#pragma unroll
  for (int t = 0; t < 32; ++t) xs[t] = -x_lds[lane + 32 * t] * L2E;

  const float Nf = (float)(wid + 1);
#pragma unroll 1
  for (int it = 0; it < 200; ++it) {
    const float mid = 0.5f * (lo + hi);
    const float c   = -mid * L2E;
    float s = 0.0f;
#pragma unroll
    for (int t = 0; t < 32; ++t) {
      const float e = __builtin_amdgcn_exp2f(xs[t] + c);   // v_exp_f32
      s += __builtin_amdgcn_rcpf(1.0f + e);                // v_rcp_f32
    }
#pragma unroll
    for (int off = 16; off; off >>= 1) s += __shfl_xor(s, off);  // wave32 reduce
    const bool too_low = s < Nf;
    lo = too_low ? mid : lo;
    hi = too_low ? hi : mid;
  }
  const float cnu = -0.5f * (lo + hi) * L2E;
#pragma unroll
  for (int t = 0; t < 32; ++t) {
    const float e = __builtin_amdgcn_exp2f(xs[t] + cnu);
    p_lds[wid][lane + 32 * t] = __builtin_amdgcn_rcpf(1.0f + e);
  }
  __syncthreads();

  // Fused attn @ v : wave w handles head w; lane owns output dims 2l, 2l+1.
  float acc0 = 0.0f, acc1 = 0.0f;
  const int d0 = 2 * lane, d1 = 2 * lane + 1;
  for (int c = 0; c < SEQ_ / 64; ++c) {
    const int j0 = c * 64;
    __syncthreads();
    for (int e = tid; e < 64 * DH_; e += 256)
      v_lds[e] = v[((size_t)b * SEQ_ + j0) * DH_ + e];
    __syncthreads();
#pragma unroll 4
    for (int jj = 0; jj < 64; ++jj) {
      const float pw = p_lds[wid][j0 + jj];
      const float pm = wid ? p_lds[wid - 1][j0 + jj] : 0.0f;
      const float wt = pw - pm;
      acc0 = fmaf(wt, v_lds[jj * DH_ + d0], acc0);
      acc1 = fmaf(wt, v_lds[jj * DH_ + d1], acc1);
    }
  }
  outh[(size_t)row * DIM_ + wid * DH_ + d0] = (__bf16)acc0;
  outh[(size_t)row * DIM_ + wid * DH_ + d1] = (__bf16)acc1;
}

// ---------------------------- output projection ----------------------------
// [4096,512] @ [512,512] + bias. 16x64 tile per wave, shared A fragment.
__global__ __launch_bounds__(256) void k_out(const __bf16* __restrict__ outh,
                                             const float*  __restrict__ wout,
                                             const float*  __restrict__ bout,
                                             float* __restrict__ out) {
  const int wid  = threadIdx.x >> 5, lane = threadIdx.x & 31;
  const int wg   = blockIdx.x * 8 + wid;        // < 2048
  const int r0   = (wg >> 3) * 16;
  const int c0   = (wg & 7) * 64;
  v8f acc[4] = {};
  for (int k0 = 0; k0 < DIM_; k0 += 32) {
    v16bf a = load_a_bf(outh, DIM_, r0, k0, lane);
#pragma unroll
    for (int t = 0; t < 4; ++t)
      acc[t] = wmma_bf16(a, load_b_f32(wout, DIM_, k0, c0 + t * 16, lane), acc[t]);
  }
  const int n = lane & 15, mb = (lane >> 4) * 8;
#pragma unroll
  for (int t = 0; t < 4; ++t) {
    const float bias = bout[c0 + t * 16 + n];
#pragma unroll
    for (int r = 0; r < 8; ++r)
      out[(size_t)(r0 + mb + r) * DIM_ + c0 + t * 16 + n] = acc[t][r] + bias;
  }
}

// ---------------------------------------------------------------------------
extern "C" void kernel_launch(void* const* d_in, const int* in_sizes, int n_in,
                              void* d_out, int out_size, void* d_ws, size_t ws_size,
                              hipStream_t stream) {
  const float* x    = (const float*)d_in[0];
  const float* g    = (const float*)d_in[1];
  const float* bta  = (const float*)d_in[2];
  const float* wqkv = (const float*)d_in[3];
  const float* wout = (const float*)d_in[4];
  const float* bout = (const float*)d_in[5];
  float* out = (float*)d_out;

  char* ws = (char*)d_ws;
  __bf16* xn   = (__bf16*)(ws);                              // 4096*512*2  = 4 MB
  __bf16* qb   = (__bf16*)(ws + (size_t)(4 << 20));          // 4096*64*2   = 512 KB
  __bf16* kb   = (__bf16*)(ws + (size_t)(4 << 20) + (512 << 10));
  float*  vf   = (float*) (ws + (size_t)(5 << 20));          // 4096*64*4   = 1 MB
  float*  dots = (float*) (ws + (size_t)(6 << 20));          // 4096*1024*4 = 16 MB
  __bf16* outh = (__bf16*)(ws + (size_t)(22 << 20));         // 4096*512*2  = 4 MB

  k_ln  <<<4096, 256, 0, stream>>>(x, g, bta, xn);
  k_qkv <<<128,  256, 0, stream>>>(xn, wqkv, qb, kb, vf);
  k_dots<<<512,  256, 0, stream>>>(qb, kb, dots);
  k_lml <<<4096, 256, 0, stream>>>(dots, vf, outh);
  k_out <<<256,  256, 0, stream>>>(outh, wout, bout, out);
}
